// MatchAttnProcessor_40424232190109
// MI455X (gfx1250) — matrix operand order repaired and checked by reading, not compiled, not run on hardware
//
#include <hip/hip_runtime.h>
#include <hip/hip_bf16.h>

// ---------------------------------------------------------------------------
// MatchAttnProcessor for MI455X (gfx1250): bf16 WMMA GEMMs + fused masked
// flash-attention (exp2-domain online softmax). Wave32.
// ---------------------------------------------------------------------------

typedef __bf16 bf16_t;
typedef __attribute__((ext_vector_type(16))) __bf16 bf16x16;
typedef __attribute__((ext_vector_type(8)))  __bf16 bf16x8;
typedef __attribute__((ext_vector_type(8)))  float  f32x8;

constexpr int Bn = 2, Ln = 1024, Sn = 2048, Cn = 1280, Hn = 8, Dn = 160;
constexpr int CORRN = 256, HALFN = Sn / 2;   // mask applies to keys [HALFN, Sn)

// log2(e)/sqrt(d): folded into Q so scores land directly in the exp2 domain.
constexpr float QSCALE = 1.4426950408889634f * 0.07905694150420949f;

// ---- WMMA operand loaders (ISA 7.12.2 layouts) ----------------------------
// A (16x32, MxK, bf16): lane m=lane&15; elems e=0..7 -> K=8*laneHi+e,
//                       e=8..15 -> K=16+8*laneHi+(e-8).
static __device__ __forceinline__ bf16x16
load_a16x32(const bf16_t* __restrict__ base, int ld, int lane) {
  const int m    = lane & 15;
  const int koff = (lane >> 4) << 3;          // 0 or 8
  const bf16_t* p = base + m * ld + koff;
  bf16x8 lo = *(const bf16x8*)(p);            // K = koff .. koff+7
  bf16x8 hi = *(const bf16x8*)(p + 16);       // K = 16+koff .. 16+koff+7
  bf16x16 r;
#pragma unroll
  for (int i = 0; i < 8; ++i) { r[i] = lo[i]; r[i + 8] = hi[i]; }
  return r;
}

// B (32x16, KxN, bf16): lane n=lane&15; elems e=0..7 -> K=8*laneHi+e,
//                       e=8..15 -> K=16+8*laneHi+(e-8).
// Memory must be N-major, K-contiguous (i.e. transposed weights / K rows / Vt).
static __device__ __forceinline__ bf16x16
load_b32x16(const bf16_t* __restrict__ base, int ld, int lane) {
  const int n    = lane & 15;
  const int koff = (lane >> 4) << 3;          // 0 or 8
  const bf16_t* p = base + n * ld + koff;
  bf16x8 lo = *(const bf16x8*)(p);
  bf16x8 hi = *(const bf16x8*)(p + 16);
  bf16x16 r;
#pragma unroll
  for (int i = 0; i < 8; ++i) { r[i] = lo[i]; r[i + 8] = hi[i]; }
  return r;
}

static __device__ __forceinline__ f32x8 wmma_bf16(bf16x16 a, bf16x16 b, f32x8 c) {
  // 8 args: (neg_a, A, neg_b, B, c_mod, C, reuse_a, reuse_b)
  return __builtin_amdgcn_wmma_f32_16x16x32_bf16(false, a, false, b, (short)0, c,
                                                 false, false);
}

// Row reductions across the 16 lanes of a C/D-layout half (xor of bits 0..3).
static __device__ __forceinline__ float rmax16(float v) {
#pragma unroll
  for (int m = 1; m <= 8; m <<= 1) v = fmaxf(v, __shfl_xor(v, m, 32));
  return v;
}
static __device__ __forceinline__ float rsum16(float v) {
#pragma unroll
  for (int m = 1; m <= 8; m <<= 1) v += __shfl_xor(v, m, 32);
  return v;
}

// ---- staging kernels ------------------------------------------------------
__global__ void cvt_bf16_kernel(const float* __restrict__ x, bf16_t* __restrict__ y, int n) {
  int i = blockIdx.x * blockDim.x + threadIdx.x;
  if (i < n) y[i] = (bf16_t)x[i];
}

// Wt[n*C + k] = W[k*C + n]  (fp32 -> bf16, transpose for WMMA-B layout)
__global__ void wtrans_kernel(const float* __restrict__ W, bf16_t* __restrict__ Wt) {
  int i = blockIdx.x * blockDim.x + threadIdx.x;
  if (i >= Cn * Cn) return;
  int n = i / Cn, k = i - n * Cn;
  Wt[i] = (bf16_t)W[k * Cn + n];
}

// ---- GEMM: Y[rows,C] = X[rows,C] @ W  (Wt is W^T in bf16) -----------------
// MODE 0: store bf16 row-major, scaled by `oscale` (Q gets log2e/sqrt(d)).
// MODE 1: scatter V^T as [B,H,D,S] bf16.
// MODE 2: fp32 out = acc + bias[col] + resid[row,col].
// Block = 128 threads (4 waves); wave w covers 64 cols at blockIdx.y*256+w*64.
template <int MODE>
__global__ __launch_bounds__(128, 1)
void gemm_kernel(const bf16_t* __restrict__ X, const bf16_t* __restrict__ Wt,
                 bf16_t* __restrict__ Ybf, float* __restrict__ Yf,
                 const float* __restrict__ bias,
                 const float* __restrict__ resid, float oscale) {
  const int lane = threadIdx.x & 31;
  const int wave = threadIdx.x >> 5;
  const int m0   = blockIdx.x * 16;
  const int n0   = blockIdx.y * 256 + wave * 64;

  f32x8 acc[4];
#pragma unroll
  for (int j = 0; j < 4; ++j)
#pragma unroll
    for (int r = 0; r < 8; ++r) acc[j][r] = 0.0f;

  for (int kb = 0; kb < Cn / 32; ++kb) {
    const int k0 = kb * 32;
    bf16x16 a = load_a16x32(X + m0 * Cn + k0, Cn, lane);
#pragma unroll
    for (int j = 0; j < 4; ++j) {
      bf16x16 bt = load_b32x16(Wt + (n0 + j * 16) * Cn + k0, Cn, lane);
      acc[j] = wmma_bf16(a, bt, acc[j]);
    }
  }

  const int mhi  = (lane >> 4) << 3;   // C/D layout: row = r + 8*laneHi
  const int nlan = lane & 15;
#pragma unroll
  for (int j = 0; j < 4; ++j) {
#pragma unroll
    for (int r = 0; r < 8; ++r) {
      const int row = m0 + mhi + r;
      const int col = n0 + j * 16 + nlan;
      const float v = acc[j][r];
      if (MODE == 0) {
        Ybf[row * Cn + col] = (bf16_t)(v * oscale);
      } else if (MODE == 1) {
        const int b = row / Sn, s = row - b * Sn;
        const int h = col / Dn, dd = col - h * Dn;
        Ybf[((b * Hn + h) * Dn + dd) * Sn + s] = (bf16_t)v;
      } else {
        Yf[row * Cn + col] = v + bias[col] + resid[row * Cn + col];
      }
    }
  }
}

// ---- fused masked flash-attention -----------------------------------------
// Block = 128 threads (4 waves); wave w owns a 16-row L tile.
// Qb: [B*L, C] bf16 (pre-scaled by log2e/sqrt(d)), Kb: [B*S, C] bf16,
// Vt: [B,H,D,S] bf16, Ob: [B*L, C] bf16.
__global__ __launch_bounds__(128, 1)
void attn_kernel(const bf16_t* __restrict__ Qb, const bf16_t* __restrict__ Kb,
                 const bf16_t* __restrict__ Vt, const int* __restrict__ seg,
                 bf16_t* __restrict__ Ob) {
  const int tid  = threadIdx.x;
  const int lane = tid & 31;
  const int wave = tid >> 5;
  const int lblk = blockIdx.x * 64;
  const int l0   = lblk + wave * 16;
  const int h    = blockIdx.y;
  const int b    = blockIdx.z;

  __shared__ int    s_corr[16][CORRN];       // corr rows lblk/4 .. lblk/4+15
  __shared__ bf16_t s_p[4][16][32];          // per-wave P staging (C->A layout)

  for (int i = tid; i < 16 * CORRN; i += 128) {
    const int rr = i >> 8, cc = i & (CORRN - 1);
    s_corr[rr][cc] = seg[b * CORRN * CORRN + ((lblk >> 2) + rr) * CORRN + cc];
  }
  __syncthreads();

  // Q tile in A layout: 5 K-chunks of 32 along d=160.
  bf16x16 qa[5];
  const bf16_t* qbase = Qb + (b * Ln + l0) * Cn + h * Dn;
#pragma unroll
  for (int t = 0; t < 5; ++t) qa[t] = load_a16x32(qbase + t * 32, Cn, lane);

  f32x8 o[10];
#pragma unroll
  for (int t = 0; t < 10; ++t)
#pragma unroll
    for (int r = 0; r < 8; ++r) o[t][r] = 0.0f;

  float mrow[8], lpart[8];                   // lpart: per-lane partial denom
#pragma unroll
  for (int r = 0; r < 8; ++r) { mrow[r] = -3.0e38f; lpart[r] = 0.0f; }

  const int   mhi   = (lane >> 4) << 3;
  const int   nlan  = lane & 15;
  const bf16_t* vhead = Vt + (b * Hn + h) * Dn * Sn;

  for (int s0 = 0; s0 < Sn; s0 += 32) {
    const bf16_t* kbase = Kb + (b * Sn + s0) * Cn + h * Dn;
    const bf16_t* vbase = vhead + s0;

    // Hint next S-block toward the caches (global_prefetch_b8).
    if (s0 + 32 < Sn) {
      __builtin_prefetch(kbase + 32 * Cn, 0, 1);
      __builtin_prefetch(vbase + 32, 0, 1);
    }

    // ---- scores (exp2 domain, Q pre-scaled): two 16x16 tiles ----
    f32x8 sc[2];
#pragma unroll
    for (int j = 0; j < 2; ++j)
#pragma unroll
      for (int r = 0; r < 8; ++r) sc[j][r] = 0.0f;

#pragma unroll
    for (int t = 0; t < 5; ++t) {
      sc[0] = wmma_bf16(qa[t], load_b32x16(kbase + t * 32, Cn, lane), sc[0]);
      sc[1] = wmma_bf16(qa[t], load_b32x16(kbase + 16 * Cn + t * 32, Cn, lane), sc[1]);
    }

    // ---- hard corr mask on second half of keys ----
#pragma unroll
    for (int j = 0; j < 2; ++j) {
      const int  ncol = s0 + j * 16 + nlan;
      const bool msk  = (s0 + j * 16) >= HALFN;   // uniform per 16-tile
      if (msk) {
#pragma unroll
        for (int r = 0; r < 8; ++r) {
          const int m = mhi + r;                  // row within this L tile
          if (s_corr[(wave << 2) + (m >> 2)][(ncol - HALFN) >> 2] != 1)
            sc[j][r] = -3.0e38f;
        }
      }
    }

    // ---- online softmax update (exp2 domain) ----
#pragma unroll
    for (int r = 0; r < 8; ++r) {
      float tmax = rmax16(fmaxf(sc[0][r], sc[1][r]));
      const float mnew = fmaxf(mrow[r], tmax);
      const float cf   = exp2f(mrow[r] - mnew);
      mrow[r] = mnew;
      const float p0 = exp2f(sc[0][r] - mnew);
      const float p1 = exp2f(sc[1][r] - mnew);
      sc[0][r] = p0; sc[1][r] = p1;
      lpart[r] = lpart[r] * cf + (p0 + p1);  // per-lane partial; reduce once at end
#pragma unroll
      for (int t = 0; t < 10; ++t) o[t][r] *= cf;
    }

    // ---- stage P through LDS: C/D layout -> A layout (in-wave transpose) ----
#pragma unroll
    for (int r = 0; r < 8; ++r) {
      s_p[wave][mhi + r][nlan]      = (bf16_t)sc[0][r];
      s_p[wave][mhi + r][16 + nlan] = (bf16_t)sc[1][r];
    }
    asm volatile("s_wait_dscnt 0" ::: "memory");   // DS in-order per wave
    bf16x16 pa;
#pragma unroll
    for (int e = 0; e < 8; ++e) {
      pa[e]     = s_p[wave][nlan][mhi + e];
      pa[e + 8] = s_p[wave][nlan][16 + mhi + e];
    }

    // ---- O += P @ V  (Vt rows are d-cols, K-contiguous in s) ----
#pragma unroll
    for (int t = 0; t < 10; ++t) {
      bf16x16 vb = load_b32x16(vbase + (t * 16) * Sn, Sn, lane);
      o[t] = wmma_bf16(pa, vb, o[t]);
    }
  }

  // ---- final denom reduction + normalize + store [B, L, H*d] bf16 ----
#pragma unroll
  for (int r = 0; r < 8; ++r) {
    const float inv = 1.0f / rsum16(lpart[r]);
    const int   row = l0 + mhi + r;
#pragma unroll
    for (int t = 0; t < 10; ++t) {
      Ob[(b * Ln + row) * Cn + h * Dn + t * 16 + nlan] = (bf16_t)(o[t][r] * inv);
    }
  }
}

// ---------------------------------------------------------------------------
extern "C" void kernel_launch(void* const* d_in, const int* in_sizes, int n_in,
                              void* d_out, int out_size, void* d_ws, size_t ws_size,
                              hipStream_t stream) {
  (void)in_sizes; (void)n_in; (void)out_size; (void)ws_size;
  const float* hs  = (const float*)d_in[0];
  const float* ehs = (const float*)d_in[1];
  const int*   seg = (const int*)d_in[2];
  const float* Wq  = (const float*)d_in[3];
  const float* Wk  = (const float*)d_in[4];
  const float* Wv  = (const float*)d_in[5];
  const float* Wo  = (const float*)d_in[6];
  const float* bo  = (const float*)d_in[7];
  float* out = (float*)d_out;

  char*  ws  = (char*)d_ws;
  size_t off = 0;
  auto carve = [&](size_t bytes) -> char* {
    char* p = ws + off; off += (bytes + 255) & ~(size_t)255; return p;
  };
  bf16_t* hsb  = (bf16_t*)carve((size_t)Bn * Ln * Cn * 2);
  bf16_t* ehsb = (bf16_t*)carve((size_t)Bn * Sn * Cn * 2);
  bf16_t* Wqt  = (bf16_t*)carve((size_t)Cn * Cn * 2);
  bf16_t* Wkt  = (bf16_t*)carve((size_t)Cn * Cn * 2);
  bf16_t* Wvt  = (bf16_t*)carve((size_t)Cn * Cn * 2);
  bf16_t* Wot  = (bf16_t*)carve((size_t)Cn * Cn * 2);
  bf16_t* Qb   = (bf16_t*)carve((size_t)Bn * Ln * Cn * 2);
  bf16_t* Kb   = (bf16_t*)carve((size_t)Bn * Sn * Cn * 2);
  bf16_t* Vtb  = (bf16_t*)carve((size_t)Bn * Hn * Dn * Sn * 2);
  bf16_t* Ob   = (bf16_t*)carve((size_t)Bn * Ln * Cn * 2);

  // Stage activations and weights in bf16 (weights transposed for WMMA-B).
  cvt_bf16_kernel<<<dim3((Bn * Ln * Cn + 255) / 256), 256, 0, stream>>>(hs, hsb, Bn * Ln * Cn);
  cvt_bf16_kernel<<<dim3((Bn * Sn * Cn + 255) / 256), 256, 0, stream>>>(ehs, ehsb, Bn * Sn * Cn);
  const int wgrid = (Cn * Cn + 255) / 256;
  wtrans_kernel<<<dim3(wgrid), 256, 0, stream>>>(Wq, Wqt);
  wtrans_kernel<<<dim3(wgrid), 256, 0, stream>>>(Wk, Wkt);
  wtrans_kernel<<<dim3(wgrid), 256, 0, stream>>>(Wv, Wvt);
  wtrans_kernel<<<dim3(wgrid), 256, 0, stream>>>(Wo, Wot);

  // Projections (Q pre-scaled into the exp2 domain).
  gemm_kernel<0><<<dim3(Bn * Ln / 16, Cn / 256), 128, 0, stream>>>(hsb,  Wqt, Qb,  nullptr, nullptr, nullptr, QSCALE);
  gemm_kernel<0><<<dim3(Bn * Sn / 16, Cn / 256), 128, 0, stream>>>(ehsb, Wkt, Kb,  nullptr, nullptr, nullptr, 1.0f);
  gemm_kernel<1><<<dim3(Bn * Sn / 16, Cn / 256), 128, 0, stream>>>(ehsb, Wvt, Vtb, nullptr, nullptr, nullptr, 1.0f);

  // Fused masked flash-attention.
  attn_kernel<<<dim3(Ln / 64, Hn, Bn), 128, 0, stream>>>(Qb, Kb, Vtb, seg, Ob);

  // Output projection + bias + residual (fp32 out).
  gemm_kernel<2><<<dim3(Bn * Ln / 16, Cn / 256), 128, 0, stream>>>(Ob, Wot, nullptr, out, bo, hs, 1.0f);
}
